// SlotAttentionCore_1099511628298
// MI455X (gfx1250) — matrix operand order repaired and checked by reading, not compiled, not run on hardware
//
#include <hip/hip_runtime.h>

// ---------------------------------------------------------------------------
// Slot-attention core for MI455X (gfx1250, wave32, WMMA bf16 16x16x32).
// All GEMMs run on v_wmma_f32_16x16x32_bf16; K/V streamed in bf16 to halve
// HBM traffic (~1.2 GB total -> ~55us at 23.3 TB/s); f32 for LN/softmax/gates.
// ---------------------------------------------------------------------------

typedef __bf16 bf16_t;
typedef __attribute__((ext_vector_type(16))) __bf16 v16bf;
typedef __attribute__((ext_vector_type(8)))  float  v8f;

#define BDIM 16
#define NSEQ 16384
#define DIMC 256
#define HEADS 4
#define DHEAD 64
#define KSLOT 8
#define EPSV  1e-8f

__device__ __forceinline__ v8f wmma_bf16(v16bf a, v16bf b, v8f c) {
  return __builtin_amdgcn_wmma_f32_16x16x32_bf16(false, a, false, b, (short)0, c,
                                                 false, false);
}

__device__ __forceinline__ v8f v8f_zero() {
  v8f z = {0.f, 0.f, 0.f, 0.f, 0.f, 0.f, 0.f, 0.f};
  return z;
}

// A-fragment (16x32 bf16): lane<16 -> row=lane, K 0..7 & 16..23; lane>=16 -> K 8..15 & 24..31
__device__ __forceinline__ v16bf frag_a_bf16(const bf16_t* base, int ld) {
  const int lane = threadIdx.x & 31;
  const bf16_t* p = base + (lane & 15) * ld + ((lane >> 4) << 3);
  v16bf f;
#pragma unroll
  for (int i = 0; i < 8; ++i) f[i] = p[i];
#pragma unroll
  for (int i = 0; i < 8; ++i) f[i + 8] = p[i + 16];
  return f;
}

__device__ __forceinline__ v16bf frag_a_f32(const float* base, int ld) {
  const int lane = threadIdx.x & 31;
  const float* p = base + (lane & 15) * ld + ((lane >> 4) << 3);
  v16bf f;
#pragma unroll
  for (int i = 0; i < 8; ++i) f[i] = (bf16_t)p[i];
#pragma unroll
  for (int i = 0; i < 8; ++i) f[i + 8] = (bf16_t)p[i + 16];
  return f;
}

// B-fragment (32x16 bf16): lane<16 -> col=lane, K 0..7 & 16..23; lane>=16 -> K 8..15 & 24..31
__device__ __forceinline__ v16bf frag_b_bf16(const bf16_t* base, int ld) {
  const int lane = threadIdx.x & 31;
  const bf16_t* p = base + (lane & 15) * ld + ((lane >> 4) << 3);
  v16bf f;
#pragma unroll
  for (int i = 0; i < 8; ++i) f[i] = p[i];
#pragma unroll
  for (int i = 0; i < 8; ++i) f[i + 8] = p[i + 16];
  return f;
}

__device__ __forceinline__ float wave_sum(float v) {
#pragma unroll
  for (int i = 16; i >= 1; i >>= 1) v += __shfl_xor(v, i, 32);
  return v;
}

// --------------------------------------------------------------------------
__global__ void f32_to_bf16_kernel(const float* __restrict__ s,
                                   bf16_t* __restrict__ d, int n) {
  int i = blockIdx.x * 256 + threadIdx.x;
  if (i < n) d[i] = (bf16_t)s[i];
}

__global__ void zero_u32_kernel(unsigned int* __restrict__ p, long long n) {
  long long i = (long long)blockIdx.x * 256 + threadIdx.x;
  long long stride = (long long)gridDim.x * 256;
  for (; i < n; i += stride) p[i] = 0u;
}

// Generic row-LN (256 cols), f32 -> bf16. One wave per row.
__global__ void __launch_bounds__(256)
ln_rows_kernel(const float* __restrict__ src, bf16_t* __restrict__ dst,
               const float* __restrict__ g, const float* __restrict__ bb, int rows) {
  const int lane = threadIdx.x & 31, wave = threadIdx.x >> 5;
  const int row = blockIdx.x * 8 + wave;
  if (row >= rows) return;
  const float* xr = src + (long long)row * DIMC;
  float vals[8];
  float s = 0.f;
#pragma unroll
  for (int e = 0; e < 8; ++e) { vals[e] = xr[lane + 32 * e]; s += vals[e]; }
  s = wave_sum(s);
  float mu = s * (1.0f / 256.0f);
  float s2 = 0.f;
#pragma unroll
  for (int e = 0; e < 8; ++e) { float d = vals[e] - mu; s2 += d * d; }
  s2 = wave_sum(s2);
  float inv = rsqrtf(s2 * (1.0f / 256.0f) + 1e-5f);
  bf16_t* dr = dst + (long long)row * DIMC;
#pragma unroll
  for (int e = 0; e < 8; ++e) {
    int c = lane + 32 * e;
    dr[c] = (bf16_t)((vals[e] - mu) * inv * g[c] + bb[c]);
  }
}

// --------------------------------------------------------------------------
// LN(inputs) fused with K/V projection. 16 rows per block; K -> [b][n][256]
// row-major bf16, V -> transposed vT[b][h][d][n] bf16.
__global__ void __launch_bounds__(256)
lnkv_kernel(const float* __restrict__ x, const bf16_t* __restrict__ wk,
            const bf16_t* __restrict__ wv, const float* __restrict__ g,
            const float* __restrict__ beta, bf16_t* __restrict__ kout,
            bf16_t* __restrict__ vT) {
  __shared__ bf16_t xs[16][264];
  const int lane = threadIdx.x & 31, wave = threadIdx.x >> 5;
  const long long gRow0 = (long long)blockIdx.x * 16;

  // LayerNorm: 2 rows per wave
  for (int rr = 0; rr < 2; ++rr) {
    int row = wave * 2 + rr;
    const float* xr = x + (gRow0 + row) * DIMC;
    float vals[8];
    float s = 0.f;
#pragma unroll
    for (int e = 0; e < 8; ++e) { vals[e] = xr[lane + 32 * e]; s += vals[e]; }
    s = wave_sum(s);
    float mu = s * (1.0f / 256.0f);
    float s2 = 0.f;
#pragma unroll
    for (int e = 0; e < 8; ++e) { float d = vals[e] - mu; s2 += d * d; }
    s2 = wave_sum(s2);
    float inv = rsqrtf(s2 * (1.0f / 256.0f) + 1e-5f);
#pragma unroll
    for (int e = 0; e < 8; ++e) {
      int c = lane + 32 * e;
      xs[row][c] = (bf16_t)((vals[e] - mu) * inv * g[c] + beta[c]);
    }
  }
  __syncthreads();

  // 32 output tiles (16 for K, 16 for V); wave handles tiles {w, w+8, w+16, w+24}
  v8f acc[4] = {v8f_zero(), v8f_zero(), v8f_zero(), v8f_zero()};
  for (int s = 0; s < 8; ++s) {
    v16bf a = frag_a_bf16(&xs[0][32 * s], 264);
#pragma unroll
    for (int t = 0; t < 4; ++t) {
      int ct = wave + 8 * t;
      const bf16_t* wb = (ct < 16) ? (wk + (size_t)(ct * 16) * DIMC + 32 * s)
                                   : (wv + (size_t)((ct - 16) * 16) * DIMC + 32 * s);
      v16bf b = frag_b_bf16(wb, DIMC);
      acc[t] = wmma_bf16(a, b, acc[t]);
    }
  }

  const int col16 = lane & 15, row0 = (lane >> 4) << 3;
  const long long bidx = gRow0 >> 14;          // / NSEQ
  const int n0 = (int)(gRow0 & (NSEQ - 1));
#pragma unroll
  for (int t = 0; t < 4; ++t) {
    int ct = wave + 8 * t;
    if (ct < 16) {
      int col = ct * 16 + col16;
#pragma unroll
      for (int r = 0; r < 8; ++r)
        kout[(gRow0 + row0 + r) * DIMC + col] = (bf16_t)acc[t][r];
    } else {
      int dcol = (ct - 16) * 16 + col16;        // 0..255 -> h,d
      long long base = ((bidx * HEADS + (dcol >> 6)) * (long long)DHEAD + (dcol & 63))
                       * (long long)NSEQ + n0 + row0;
#pragma unroll
      for (int r = 0; r < 8; ++r) vT[base + r] = (bf16_t)acc[t][r];
    }
  }
}

// colsum of V over n: vsum[(b*H+h)*64+d]
__global__ void __launch_bounds__(256)
vsum_kernel(const bf16_t* __restrict__ vT, float* __restrict__ vsum) {
  const int lane = threadIdx.x & 31, wave = threadIdx.x >> 5;
  const int row = blockIdx.x * 8 + wave;        // 0..4095
  const bf16_t* p = vT + (long long)row * NSEQ;
  float s = 0.f;
  for (int e = 0; e < NSEQ / 32; ++e) s += (float)p[lane + 32 * e];
  s = wave_sum(s);
  if (lane == 0) vsum[row] = s;
}

// q = ln(slots) @ w_q.T -> qpad[(b*H+h)][16][64] (rows 8..15 stay zero)
__global__ void __launch_bounds__(256)
qproj_kernel(const bf16_t* __restrict__ xln, const bf16_t* __restrict__ wq,
             bf16_t* __restrict__ qpad) {
  const int lane = threadIdx.x & 31, wave = threadIdx.x >> 5;
  const int tile = blockIdx.x * 8 + wave;       // 0..127
  const int mt = tile >> 4, ct = tile & 15;
  v8f acc = v8f_zero();
  for (int s = 0; s < 8; ++s) {
    v16bf a = frag_a_bf16(xln + (size_t)(mt * 16) * DIMC + 32 * s, DIMC);
    v16bf b = frag_b_bf16(wq + (size_t)(ct * 16) * DIMC + 32 * s, DIMC);
    acc = wmma_bf16(a, b, acc);
  }
  const int c = ct * 16 + (lane & 15);
  const int h = c >> 6, d = c & 63;
  const int row0 = (lane >> 4) << 3;
#pragma unroll
  for (int r = 0; r < 8; ++r) {
    int m = mt * 16 + row0 + r;                 // 0..127
    int bb = m >> 3, i = m & 7;
    qpad[(((bb * HEADS + h) * 16) + i) * DHEAD + d] = (bf16_t)acc[r];
  }
}

// dots[b][i*H+h][j] = scale * q . k   (one 16x16 j-tile per wave)
__global__ void __launch_bounds__(256)
dots_kernel(const bf16_t* __restrict__ qpad, const bf16_t* __restrict__ kbuf,
            float* __restrict__ dots) {
  const int lane = threadIdx.x & 31, wave = threadIdx.x >> 5;
  const int bh = blockIdx.y, b = bh >> 2, h = bh & 3;
  const int j0 = blockIdx.x * 128 + wave * 16;
  const bf16_t* qb = qpad + (size_t)bh * 16 * DHEAD;
  v16bf a0 = frag_a_bf16(qb, DHEAD);
  v16bf a1 = frag_a_bf16(qb + 32, DHEAD);
  const bf16_t* kb = kbuf + ((long long)b * NSEQ + j0) * DIMC + h * DHEAD;
  v16bf b0 = frag_b_bf16(kb, DIMC);
  v16bf b1 = frag_b_bf16(kb + 32, DIMC);
  v8f acc = v8f_zero();
  acc = wmma_bf16(a0, b0, acc);
  acc = wmma_bf16(a1, b1, acc);
  const int j = j0 + (lane & 15), row0 = (lane >> 4) << 3;
#pragma unroll
  for (int r = 0; r < 8; ++r) {
    int i = row0 + r;
    if (i < 8)
      dots[((long long)(b * 32 + i * HEADS + h)) * NSEQ + j] = acc[r] * 0.125f; // DH^-0.5
  }
}

// joint softmax over the 32 (slot,head) rows per column + attn_out + row sums
__global__ void __launch_bounds__(256)
softmax_kernel(const float* __restrict__ dots, bf16_t* __restrict__ attnb,
               float* __restrict__ attn_out, float* __restrict__ srow) {
  __shared__ float ssum[32];
  const int b = blockIdx.y;
  const int j = blockIdx.x * 256 + threadIdx.x;
  if (threadIdx.x < 32) ssum[threadIdx.x] = 0.f;
  __syncthreads();
  float v[32];
  const float* dp = dots + (long long)b * 32 * NSEQ + j;
  float mx = -3.4e38f;
#pragma unroll
  for (int r = 0; r < 32; ++r) { v[r] = dp[(long long)r * NSEQ]; mx = fmaxf(mx, v[r]); }
  float s = 0.f;
#pragma unroll
  for (int r = 0; r < 32; ++r) { v[r] = expf(v[r] - mx); s += v[r]; }
  float inv = 1.f / s;
  bf16_t* ap = attnb + (long long)b * 64 * NSEQ + j;
#pragma unroll
  for (int r = 0; r < 32; ++r) {
    float a = v[r] * inv;
    ap[(long long)r * NSEQ] = (bf16_t)a;
    atomicAdd(&ssum[r], a);
  }
  float* op = attn_out + (long long)b * KSLOT * NSEQ + j;
#pragma unroll
  for (int i = 0; i < 8; ++i)
    op[(long long)i * NSEQ] =
        (v[4 * i] + v[4 * i + 1] + v[4 * i + 2] + v[4 * i + 3]) * inv * 0.25f;
  __syncthreads();
  if (threadIdx.x < 32) atomicAdd(&srow[b * 32 + threadIdx.x], ssum[threadIdx.x]);
}

// updates[b*8+i][h*64+d] = (attn@V + eps*vsum) / (S + eps). One block per (b,h),
// 8 waves split the j-range, LDS cross-wave reduction.
__global__ void __launch_bounds__(256)
updates_kernel(const bf16_t* __restrict__ attnb, const bf16_t* __restrict__ vT,
               const float* __restrict__ vsum, const float* __restrict__ srow,
               float* __restrict__ upd) {
  __shared__ float red[8][16][64];
  const int lane = threadIdx.x & 31, wave = threadIdx.x >> 5;
  const int bh = blockIdx.x, b = bh >> 2, h = bh & 3;
  const bf16_t* abase = attnb + ((long long)(b * 64 + h)) * NSEQ; // row i at + i*4*NSEQ
  const bf16_t* vb = vT + (long long)bh * DHEAD * NSEQ;
  v8f acc[4] = {v8f_zero(), v8f_zero(), v8f_zero(), v8f_zero()};
  const int jb0 = wave * (NSEQ / 8);
  for (int s = 0; s < NSEQ / 8 / 32; ++s) {
    int j = jb0 + s * 32;
    v16bf a = frag_a_bf16(abase + j, 4 * NSEQ);
#pragma unroll
    for (int t = 0; t < 4; ++t) {
      v16bf bb = frag_b_bf16(vb + (long long)(t * 16) * NSEQ + j, NSEQ);
      acc[t] = wmma_bf16(a, bb, acc[t]);
    }
  }
  const int col16 = lane & 15, row0 = (lane >> 4) << 3;
#pragma unroll
  for (int t = 0; t < 4; ++t)
#pragma unroll
    for (int r = 0; r < 8; ++r) red[wave][row0 + r][t * 16 + col16] = acc[t][r];
  __syncthreads();
  for (int e = threadIdx.x; e < 1024; e += 256) {
    int i = e >> 6, d = e & 63;
    float s = 0.f;
#pragma unroll
    for (int w = 0; w < 8; ++w) s += red[w][i][d];
    if (i < 8) {
      float num = s + EPSV * vsum[bh * DHEAD + d];
      float den = srow[b * 32 + i * HEADS + h] + EPSV;
      upd[(long long)(b * KSLOT + i) * DIMC + h * DHEAD + d] = num / den;
    }
  }
}

// GRU cell: 6 WMMA accumulators per 16x16 output tile (gi/gh x r/z/n)
__global__ void __launch_bounds__(256)
gru_kernel(const float* __restrict__ upd, const float* __restrict__ hprev,
           const bf16_t* __restrict__ wih, const bf16_t* __restrict__ whh,
           const float* __restrict__ bih, const float* __restrict__ bhh,
           float* __restrict__ hout) {
  const int lane = threadIdx.x & 31, wave = threadIdx.x >> 5;
  const int tile = blockIdx.x * 8 + wave;       // 0..127
  const int mt = tile >> 4, ct = tile & 15;
  v8f gir = v8f_zero(), giz = v8f_zero(), gin = v8f_zero();
  v8f ghr = v8f_zero(), ghz = v8f_zero(), ghn = v8f_zero();
  for (int s = 0; s < 8; ++s) {
    v16bf au = frag_a_f32(upd + (size_t)(mt * 16) * DIMC + 32 * s, DIMC);
    v16bf ah = frag_a_f32(hprev + (size_t)(mt * 16) * DIMC + 32 * s, DIMC);
    v16bf b0 = frag_b_bf16(wih + (size_t)(0 * 256 + ct * 16) * DIMC + 32 * s, DIMC);
    v16bf b1 = frag_b_bf16(wih + (size_t)(1 * 256 + ct * 16) * DIMC + 32 * s, DIMC);
    v16bf b2 = frag_b_bf16(wih + (size_t)(2 * 256 + ct * 16) * DIMC + 32 * s, DIMC);
    gir = wmma_bf16(au, b0, gir);
    giz = wmma_bf16(au, b1, giz);
    gin = wmma_bf16(au, b2, gin);
    b0 = frag_b_bf16(whh + (size_t)(0 * 256 + ct * 16) * DIMC + 32 * s, DIMC);
    b1 = frag_b_bf16(whh + (size_t)(1 * 256 + ct * 16) * DIMC + 32 * s, DIMC);
    b2 = frag_b_bf16(whh + (size_t)(2 * 256 + ct * 16) * DIMC + 32 * s, DIMC);
    ghr = wmma_bf16(ah, b0, ghr);
    ghz = wmma_bf16(ah, b1, ghz);
    ghn = wmma_bf16(ah, b2, ghn);
  }
  const int c = ct * 16 + (lane & 15);
  const float br_ = bih[c], bz_ = bih[256 + c], bn_ = bih[512 + c];
  const float cr_ = bhh[c], cz_ = bhh[256 + c], cn_ = bhh[512 + c];
  const int row0 = (lane >> 4) << 3;
#pragma unroll
  for (int r = 0; r < 8; ++r) {
    int m = mt * 16 + row0 + r;
    float hv = hprev[(size_t)m * DIMC + c];
    float rg = 1.f / (1.f + expf(-(gir[r] + br_ + ghr[r] + cr_)));
    float zg = 1.f / (1.f + expf(-(giz[r] + bz_ + ghz[r] + cz_)));
    float ng = tanhf(gin[r] + bn_ + rg * (ghn[r] + cn_));
    hout[(size_t)m * DIMC + c] = (1.f - zg) * ng + zg * hv;
  }
}

// FFN up-proj + exact GELU -> bf16 [128,1024]
__global__ void __launch_bounds__(256)
ff1_kernel(const bf16_t* __restrict__ xln, const bf16_t* __restrict__ w1,
           const float* __restrict__ b1, bf16_t* __restrict__ out) {
  const int lane = threadIdx.x & 31, wave = threadIdx.x >> 5;
  const int tile = blockIdx.x * 8 + wave;       // 0..511
  const int mt = tile >> 6, ct = tile & 63;
  v8f acc = v8f_zero();
  for (int s = 0; s < 8; ++s) {
    v16bf a = frag_a_bf16(xln + (size_t)(mt * 16) * DIMC + 32 * s, DIMC);
    v16bf b = frag_b_bf16(w1 + (size_t)(ct * 16) * DIMC + 32 * s, DIMC);
    acc = wmma_bf16(a, b, acc);
  }
  const int c = ct * 16 + (lane & 15);
  const float bias = b1[c];
  const int row0 = (lane >> 4) << 3;
#pragma unroll
  for (int r = 0; r < 8; ++r) {
    int m = mt * 16 + row0 + r;
    float x = acc[r] + bias;
    float ge = 0.5f * x * (1.f + erff(x * 0.70710678f));
    out[(size_t)m * 1024 + c] = (bf16_t)ge;
  }
}

// FFN down-proj + residual; writes slots buffer AND d_out slots region
__global__ void __launch_bounds__(256)
ff2_kernel(const bf16_t* __restrict__ ffh, const bf16_t* __restrict__ w2,
           const float* __restrict__ b2, const float* __restrict__ mid,
           float* __restrict__ slots, float* __restrict__ outslots) {
  const int lane = threadIdx.x & 31, wave = threadIdx.x >> 5;
  const int tile = blockIdx.x * 8 + wave;       // 0..127
  const int mt = tile >> 4, ct = tile & 15;
  v8f acc = v8f_zero();
  for (int s = 0; s < 32; ++s) {
    v16bf a = frag_a_bf16(ffh + (size_t)(mt * 16) * 1024 + 32 * s, 1024);
    v16bf b = frag_b_bf16(w2 + (size_t)(ct * 16) * 1024 + 32 * s, 1024);
    acc = wmma_bf16(a, b, acc);
  }
  const int c = ct * 16 + (lane & 15);
  const float bias = b2[c];
  const int row0 = (lane >> 4) << 3;
#pragma unroll
  for (int r = 0; r < 8; ++r) {
    int m = mt * 16 + row0 + r;
    float v = mid[(size_t)m * DIMC + c] + acc[r] + bias;
    slots[(size_t)m * DIMC + c] = v;
    outslots[(size_t)m * DIMC + c] = v;
  }
}

// --------------------------------------------------------------------------
extern "C" void kernel_launch(void* const* d_in, const int* in_sizes, int n_in,
                              void* d_out, int out_size, void* d_ws, size_t ws_size,
                              hipStream_t stream) {
  const float* inputs  = (const float*)d_in[0];
  const float* slots_i = (const float*)d_in[1];
  const float* w_q  = (const float*)d_in[2];
  const float* w_k  = (const float*)d_in[3];
  const float* w_v  = (const float*)d_in[4];
  const float* w_ih = (const float*)d_in[5];
  const float* w_hh = (const float*)d_in[6];
  const float* b_ih = (const float*)d_in[7];
  const float* b_hh = (const float*)d_in[8];
  const float* ln_in_g = (const float*)d_in[9];
  const float* ln_in_b = (const float*)d_in[10];
  const float* ln_s_g  = (const float*)d_in[11];
  const float* ln_s_b  = (const float*)d_in[12];
  const float* ln_ff_g = (const float*)d_in[13];
  const float* ln_ff_b = (const float*)d_in[14];
  const float* w_ff1 = (const float*)d_in[15];
  const float* b_ff1 = (const float*)d_in[16];
  const float* w_ff2 = (const float*)d_in[17];
  const float* b_ff2 = (const float*)d_in[18];

  char* p = (char*)d_ws;
  auto alloc = [&](size_t bytes) {
    char* r = p;
    p += (bytes + 255) & ~(size_t)255;
    return r;
  };
  const long long BN = (long long)BDIM * NSEQ;
  bf16_t* wq_bf  = (bf16_t*)alloc((size_t)256 * 256 * 2);
  bf16_t* wk_bf  = (bf16_t*)alloc((size_t)256 * 256 * 2);
  bf16_t* wv_bf  = (bf16_t*)alloc((size_t)256 * 256 * 2);
  bf16_t* wih_bf = (bf16_t*)alloc((size_t)768 * 256 * 2);
  bf16_t* whh_bf = (bf16_t*)alloc((size_t)768 * 256 * 2);
  bf16_t* wf1_bf = (bf16_t*)alloc((size_t)1024 * 256 * 2);
  bf16_t* wf2_bf = (bf16_t*)alloc((size_t)256 * 1024 * 2);
  bf16_t* kbuf   = (bf16_t*)alloc((size_t)BN * DIMC * 2);
  bf16_t* vT     = (bf16_t*)alloc((size_t)BN * DIMC * 2);
  float*  vsum   = (float*) alloc((size_t)BDIM * HEADS * DHEAD * 4);
  bf16_t* qpad   = (bf16_t*)alloc((size_t)BDIM * HEADS * 16 * DHEAD * 2);
  float*  dots   = (float*) alloc((size_t)BDIM * 32 * NSEQ * 4);
  bf16_t* attnb  = (bf16_t*)alloc((size_t)BDIM * 64 * NSEQ * 2);
  float*  srow   = (float*) alloc((size_t)BDIM * 32 * 4);
  float*  slotsb = (float*) alloc((size_t)128 * 256 * 4);
  float*  slotsm = (float*) alloc((size_t)128 * 256 * 4);
  bf16_t* slotsl = (bf16_t*)alloc((size_t)128 * 256 * 2);
  float*  upd    = (float*) alloc((size_t)128 * 256 * 4);
  bf16_t* ffln   = (bf16_t*)alloc((size_t)128 * 256 * 2);
  bf16_t* ffh    = (bf16_t*)alloc((size_t)128 * 1024 * 2);

  // Weights -> bf16 once per launch
  f32_to_bf16_kernel<<<(65536 + 255) / 256, 256, 0, stream>>>(w_q, wq_bf, 65536);
  f32_to_bf16_kernel<<<(65536 + 255) / 256, 256, 0, stream>>>(w_k, wk_bf, 65536);
  f32_to_bf16_kernel<<<(65536 + 255) / 256, 256, 0, stream>>>(w_v, wv_bf, 65536);
  f32_to_bf16_kernel<<<(196608 + 255) / 256, 256, 0, stream>>>(w_ih, wih_bf, 196608);
  f32_to_bf16_kernel<<<(196608 + 255) / 256, 256, 0, stream>>>(w_hh, whh_bf, 196608);
  f32_to_bf16_kernel<<<(262144 + 255) / 256, 256, 0, stream>>>(w_ff1, wf1_bf, 262144);
  f32_to_bf16_kernel<<<(262144 + 255) / 256, 256, 0, stream>>>(w_ff2, wf2_bf, 262144);
  // zero pads (attn rows 32..63 and qpad rows 8..15 must stay zero)
  zero_u32_kernel<<<4096, 256, 0, stream>>>((unsigned int*)attnb,
                                            (long long)BDIM * 64 * NSEQ / 2);
  zero_u32_kernel<<<128, 256, 0, stream>>>((unsigned int*)qpad,
                                           (long long)BDIM * HEADS * 16 * DHEAD / 2);
  hipMemcpyAsync(slotsb, slots_i, (size_t)128 * 256 * 4, hipMemcpyDeviceToDevice, stream);

  // LN(inputs) + K/V projection (bulk of FLOPs), then V column sums
  lnkv_kernel<<<(int)(BN / 16), 256, 0, stream>>>(inputs, wk_bf, wv_bf, ln_in_g,
                                                  ln_in_b, kbuf, vT);
  vsum_kernel<<<BDIM * HEADS * DHEAD / 8, 256, 0, stream>>>(vT, vsum);

  float* attn_out = (float*)d_out + 128 * 256;
  for (int it = 0; it < 3; ++it) {
    ln_rows_kernel<<<16, 256, 0, stream>>>(slotsb, slotsl, ln_s_g, ln_s_b, 128);
    qproj_kernel<<<16, 256, 0, stream>>>(slotsl, wq_bf, qpad);
    zero_u32_kernel<<<1, 256, 0, stream>>>((unsigned int*)srow, BDIM * 32);
    dots_kernel<<<dim3(NSEQ / 128, BDIM * HEADS), 256, 0, stream>>>(qpad, kbuf, dots);
    softmax_kernel<<<dim3(NSEQ / 256, BDIM), 256, 0, stream>>>(dots, attnb, attn_out,
                                                               srow);
    updates_kernel<<<BDIM * HEADS, 256, 0, stream>>>(attnb, vT, vsum, srow, upd);
    gru_kernel<<<16, 256, 0, stream>>>(upd, slotsb, wih_bf, whh_bf, b_ih, b_hh, slotsm);
    ln_rows_kernel<<<16, 256, 0, stream>>>(slotsm, ffln, ln_ff_g, ln_ff_b, 128);
    ff1_kernel<<<64, 256, 0, stream>>>(ffln, wf1_bf, b_ff1, ffh);
    ff2_kernel<<<16, 256, 0, stream>>>(ffh, wf2_bf, b_ff2, slotsm, slotsb,
                                       (float*)d_out);
  }
}
